// Net_40355512714135
// MI455X (gfx1250) — matrix operand order repaired and checked by reading, not compiled, not run on hardware
//
#include <hip/hip_runtime.h>
#include <hip/hip_bf16.h>

typedef __attribute__((ext_vector_type(16))) __bf16 v16bf;
typedef __attribute__((ext_vector_type(8)))  float  v8f;

#define B_   64
#define T_   256
#define D_   64
#define H_   1024
#define G3H_ (3 * H_)
#define BT_  (B_ * T_)
#define NBLK (H_ / 16)   // 64 persistent blocks per layer

// ---------- helpers ----------

__device__ __forceinline__ unsigned short f2bf(float f) {
  unsigned int u = __float_as_uint(f);
  u += 0x7FFFu + ((u >> 16) & 1u);            // round-to-nearest-even
  return (unsigned short)(u >> 16);
}

__device__ __forceinline__ float sigm_(float x) {
  x = fminf(fmaxf(x, -30.f), 30.f);
  return 1.f / (1.f + __expf(-x));
}
__device__ __forceinline__ float tanh_(float x) {
  x = fminf(fmaxf(x, -15.f), 15.f);
  float e = __expf(2.f * x);
  return (e - 1.f) / (e + 1.f);
}

// CDNA5 async copy: 16B global -> LDS, tracked by ASYNCcnt (ISA 08 §4).
__device__ __forceinline__ void async_g2l_16B(unsigned lds_off, const void* gaddr) {
  asm volatile("global_load_async_to_lds_b128 %0, %1, off"
               :: "v"(lds_off), "v"(gaddr) : "memory");
}
__device__ __forceinline__ void wait_async0() {
  asm volatile("s_wait_asynccnt 0" ::: "memory");
}

// Device-wide barrier: atomic arrive + generation flag, s_sleep spin.
// Safe here: 64 blocks x 140KB LDS = 2 blocks/WGP -> co-resident on MI455X.
__device__ __forceinline__ void grid_sync(unsigned* cnt, unsigned* gen, int nblocks) {
  __syncthreads();
  if (threadIdx.x == 0) {
    __threadfence();   // make this block's stores agent-visible (release)
    unsigned g = __hip_atomic_load(gen, __ATOMIC_RELAXED, __HIP_MEMORY_SCOPE_AGENT);
    unsigned a = __hip_atomic_fetch_add(cnt, 1u, __ATOMIC_ACQ_REL, __HIP_MEMORY_SCOPE_AGENT);
    if (a == (unsigned)(nblocks - 1)) {
      __hip_atomic_store(cnt, 0u, __ATOMIC_RELAXED, __HIP_MEMORY_SCOPE_AGENT);
      __hip_atomic_store(gen, g + 1u, __ATOMIC_RELEASE, __HIP_MEMORY_SCOPE_AGENT);
    } else {
      while (__hip_atomic_load(gen, __ATOMIC_ACQUIRE, __HIP_MEMORY_SCOPE_AGENT) == g)
        __builtin_amdgcn_s_sleep(8);
    }
    __threadfence();   // acquire side: invalidate stale near caches
  }
  __syncthreads();
}

// A-operand (16x32 bf16) packed layout for [M,K] row-major source (ISA 7.12.2):
// lane = m + 16*((kk>>3)&1), slot = (kk&7) + 8*(kk>>4); 16 bf16 per lane contiguous.
__device__ __forceinline__ size_t a_idx(int row, int col, int KT) {
  int mt = row >> 4, m = row & 15;
  int kt = col >> 5, kk = col & 31;
  int lane = m + (((kk >> 3) & 1) << 4);
  int s = (kk & 7) + ((kk >> 4) << 3);
  return ((size_t)(mt * KT + kt) * 32 + lane) * 16 + s;
}

// B-operand (32x16 bf16) packed layout; B[k][n] = W[n][k] for weight W [N,K].
__device__ __forceinline__ size_t b_idx(int k, int n, int KT) {
  int nt = n >> 4, nn = n & 15;
  int kt = k >> 5, kk = k & 31;
  int lane = nn + (((kk >> 3) & 1) << 4);
  int s = (kk & 7) + ((kk >> 4) << 3);
  return ((size_t)(nt * KT + kt) * 32 + lane) * 16 + s;
}

// ---------- packing kernels ----------

__global__ void pack_a_f32(const float* __restrict__ src, unsigned short* __restrict__ dst,
                           int M, int K) {
  size_t i = (size_t)blockIdx.x * blockDim.x + threadIdx.x;
  if (i >= (size_t)M * K) return;
  int row = (int)(i / K), col = (int)(i % K);
  dst[a_idx(row, col, K >> 5)] = f2bf(src[i]);
}

__global__ void pack_b_w(const float* __restrict__ w, unsigned short* __restrict__ dst,
                         int N, int K) {
  size_t i = (size_t)blockIdx.x * blockDim.x + threadIdx.x;
  if (i >= (size_t)N * K) return;
  int n = (int)(i / K), k = (int)(i % K);
  dst[b_idx(k, n, K >> 5)] = f2bf(w[i]);
}

__global__ void zero_u32(unsigned int* __restrict__ p, int n) {
  int i = blockIdx.x * blockDim.x + threadIdx.x;
  if (i < n) p[i] = 0u;
}

// ---------- big WMMA GEMM: C[M,N] = Apacked x Bpacked ----------
// grid = (N/16, M/64); block = 128 (4 waves). The 16-wide B column strip
// (KT*1KB <= 32KB) is staged into LDS once per block via async-to-LDS and
// shared by all 4 waves; A streams from global, software-pipelined 1 K-tile.

__global__ void __launch_bounds__(128)
wmma_gemm(const unsigned short* __restrict__ pa, const unsigned short* __restrict__ pb,
          float* __restrict__ c, int N, int KT) {
  __shared__ unsigned short lds_b[16 * 1024];     // 32 KB max (KT <= 32)
  const int tid  = threadIdx.x;
  const int lane = tid & 31;
  const int w    = tid >> 5;
  const int nt   = blockIdx.x;

  { // stage B strip: KT*512 ushorts = KT*64 x 16B chunks
    const char* gsrc = (const char*)(pb + (size_t)nt * KT * 512);
    const unsigned lbase = (unsigned)(size_t)(&lds_b[0]);
    const int chunks = KT * 64;
    for (int ck = tid; ck < chunks; ck += 128)
      async_g2l_16B(lbase + ck * 16, gsrc + (size_t)ck * 16);
    wait_async0();
    __syncthreads();
  }

  const int mt = blockIdx.y * 4 + w;
  const v16bf* ap = reinterpret_cast<const v16bf*>(pa) + (size_t)mt * KT * 32 + lane;
  v16bf acur = ap[0];
  v8f acc = {};
  for (int kt = 0; kt < KT; ++kt) {
    const v16bf* bsh = reinterpret_cast<const v16bf*>(lds_b + ((size_t)kt * 32 + lane) * 16);
    v16bf b = *bsh;                               // ds_load_b128 x2
    const int ktn = (kt + 1 < KT) ? kt + 1 : kt;
    if (kt + 4 < KT) __builtin_prefetch((const void*)(ap + (size_t)(kt + 4) * 32), 0, 3);
    v16bf anxt = ap[(size_t)ktn * 32];            // in flight during wmma
    acc = __builtin_amdgcn_wmma_f32_16x16x32_bf16(false, acur, false, b, (short)0, acc,
                                                  false, false);
    acur = anxt;
  }
  const int col  = nt * 16 + (lane & 15);
  const int rowb = mt * 16 + ((lane >> 4) << 3);
  #pragma unroll
  for (int v = 0; v < 8; ++v)
    c[(size_t)(rowb + v) * N + col] = acc[v];
}

// ---------- persistent GRU layer: all 256 timesteps in one launch ----------
// 64 blocks (one 16-wide H slice each), 384 threads = 12 waves.
// Per step:
//  1: async-stage the whole packed h state (128 KB) into LDS.
//  2: wave (gate, mtile) does 32 WMMAs: A from LDS, B (w_hh, L2-hot) from
//     global pipelined one K-tile ahead; hg -> LDS.
//  3: fused gate math, h update, bf16 repack (double-buffered state + seq out).
//  4: device-wide barrier (replaces 256 kernel launches).

__global__ void __launch_bounds__(384)
gru_layer(const float* __restrict__ xg,              // [BT, 3H]
          unsigned short* __restrict__ hpA,          // A-packed [64,1024] bf16 (buf 0)
          unsigned short* __restrict__ hpB,          // A-packed [64,1024] bf16 (buf 1)
          float* __restrict__ hf32,                  // [64,1024] fp32 state
          const unsigned short* __restrict__ wpack,  // B-packed [1024 x 3072] bf16
          const float* __restrict__ b_ih, const float* __restrict__ b_hh,
          unsigned short* __restrict__ seqpack,      // A-packed [BT,1024] or nullptr
          unsigned* __restrict__ bar) {              // [0]=cnt [1]=gen (cnt pre-zeroed)
  __shared__ unsigned short lds_h[B_ * H_];          // 128 KB packed h state
  __shared__ float hg[3][B_][16];                    // 12 KB gate pre-activations
  const int tid  = threadIdx.x;
  const int lane = tid & 31;
  const int wid  = tid >> 5;                         // 0..11
  const int KT   = H_ >> 5;                          // 32
  const int mt   = wid & 3;
  const int gate = wid >> 2;
  const int nt   = gate * (H_ >> 4) + blockIdx.x;
  const v16bf* bp = reinterpret_cast<const v16bf*>(wpack) + (size_t)nt * KT * 32 + lane;
  const unsigned lbase = (unsigned)(size_t)(&lds_h[0]);

  for (int t = 0; t < T_; ++t) {
    const unsigned short* hin = (t & 1) ? hpB : hpA;
    unsigned short*      hout = (t & 1) ? hpA : hpB;

    { // phase 1: 8192 x 16B async chunks
      const char* gsrc = (const char*)hin;
      for (int ck = tid; ck < (B_ * H_ * 2) / 16; ck += 384)
        async_g2l_16B(lbase + ck * 16, gsrc + (size_t)ck * 16);
      wait_async0();
      __syncthreads();
    }

    { // phase 2
      v16bf bcur = bp[0];
      v8f acc = {};
      for (int kt = 0; kt < KT; ++kt) {
        const v16bf* ash =
            reinterpret_cast<const v16bf*>(lds_h + ((size_t)(mt * KT + kt) * 32 + lane) * 16);
        v16bf a = *ash;                              // ds_load_b128 x2
        const int ktn = (kt + 1 < KT) ? kt + 1 : kt;
        v16bf bnxt = bp[(size_t)ktn * 32];           // overlaps the wmma below
        acc = __builtin_amdgcn_wmma_f32_16x16x32_bf16(false, a, false, bcur, (short)0, acc,
                                                      false, false);
        bcur = bnxt;
      }
      const int jj   = lane & 15;
      const int rowb = mt * 16 + ((lane >> 4) << 3);
      #pragma unroll
      for (int v = 0; v < 8; ++v) hg[gate][rowb + v][jj] = acc[v];
    }
    __syncthreads();

    // phase 3
    for (int e = tid; e < B_ * 16; e += 384) {
      const int b   = e >> 4;
      const int jj  = e & 15;
      const int col = blockIdx.x * 16 + jj;
      const float* xrow = xg + (size_t)(b * T_ + t) * G3H_;
      const float xr = xrow[col]          + b_ih[col];
      const float xz = xrow[H_ + col]     + b_ih[H_ + col];
      const float xn = xrow[2 * H_ + col] + b_ih[2 * H_ + col];
      const float hr = hg[0][b][jj] + b_hh[col];
      const float hz = hg[1][b][jj] + b_hh[H_ + col];
      const float hn = hg[2][b][jj] + b_hh[2 * H_ + col];
      const float r = sigm_(xr + hr);
      const float z = sigm_(xz + hz);
      const float n = tanh_(xn + r * hn);
      const size_t hidx = (size_t)b * H_ + col;
      const float hnew = (1.f - z) * n + z * hf32[hidx];
      hf32[hidx] = hnew;
      const unsigned short hb = f2bf(hnew);
      hout[a_idx(b, col, KT)] = hb;
      if (seqpack) seqpack[a_idx(b * T_ + t, col, KT)] = hb;
    }

    // phase 4: one device-wide barrier per step (double buffering makes
    // a single sync sufficient: t reads hin before it, t+1 writes it after).
    grid_sync(&bar[0], &bar[1], NBLK);
  }
}

// ---------- final FC: out[b] = h[b,:] . fc_w + fc_b ----------

__global__ void __launch_bounds__(128)
fc_kernel(const float* __restrict__ h, const float* __restrict__ w,
          const float* __restrict__ bias, float* __restrict__ out) {
  __shared__ float red[128];
  const int b = blockIdx.x;
  float s = 0.f;
  for (int j = threadIdx.x; j < H_; j += 128) s += h[(size_t)b * H_ + j] * w[j];
  red[threadIdx.x] = s;
  __syncthreads();
  for (int off = 64; off > 0; off >>= 1) {
    if (threadIdx.x < off) red[threadIdx.x] += red[threadIdx.x + off];
    __syncthreads();
  }
  if (threadIdx.x == 0) out[b] = red[0] + bias[0];
}

// ---------- launch ----------

extern "C" void kernel_launch(void* const* d_in, const int* in_sizes, int n_in,
                              void* d_out, int out_size, void* d_ws, size_t ws_size,
                              hipStream_t stream) {
  (void)in_sizes; (void)n_in; (void)out_size; (void)ws_size;
  const float* x      = (const float*)d_in[0];
  const float* w_ih0  = (const float*)d_in[1];
  const float* w_hh0  = (const float*)d_in[2];
  const float* b_ih0  = (const float*)d_in[3];
  const float* b_hh0  = (const float*)d_in[4];
  const float* w_ih1  = (const float*)d_in[5];
  const float* w_hh1  = (const float*)d_in[6];
  const float* b_ih1  = (const float*)d_in[7];
  const float* b_hh1  = (const float*)d_in[8];
  const float* fc_w   = (const float*)d_in[9];
  const float* fc_b   = (const float*)d_in[10];
  float* out = (float*)d_out;

  char* base = (char*)d_ws;
  size_t off = 0;
  auto alloc = [&](size_t bytes) { size_t o = off; off += (bytes + 255) & ~(size_t)255; return o; };
  const size_t oXG    = alloc((size_t)BT_ * G3H_ * 4);   // 201 MB, reused for both layers
  const size_t oXPK   = alloc((size_t)BT_ * D_ * 2);
  const size_t oH1PK  = alloc((size_t)BT_ * H_ * 2);     // layer-0 output, A-packed
  const size_t oWHH0  = alloc((size_t)G3H_ * H_ * 2);
  const size_t oWHH1  = alloc((size_t)G3H_ * H_ * 2);
  const size_t oWIH0  = alloc((size_t)G3H_ * D_ * 2);
  const size_t oWIH1  = alloc((size_t)G3H_ * H_ * 2);
  const size_t oHF32  = alloc((size_t)B_ * H_ * 4);
  const size_t oHP0   = alloc((size_t)B_ * H_ * 2);
  const size_t oHP1   = alloc((size_t)B_ * H_ * 2);
  const size_t oBAR   = alloc(256);

  float*          xg    = (float*)(base + oXG);
  unsigned short* xpk   = (unsigned short*)(base + oXPK);
  unsigned short* h1pk  = (unsigned short*)(base + oH1PK);
  unsigned short* whh0p = (unsigned short*)(base + oWHH0);
  unsigned short* whh1p = (unsigned short*)(base + oWHH1);
  unsigned short* wih0p = (unsigned short*)(base + oWIH0);
  unsigned short* wih1p = (unsigned short*)(base + oWIH1);
  float*          hf32  = (float*)(base + oHF32);
  unsigned short* hp0   = (unsigned short*)(base + oHP0);
  unsigned short* hp1   = (unsigned short*)(base + oHP1);
  unsigned*       bar   = (unsigned*)(base + oBAR);

  // 1) pack weights (bf16 B-fragments) and x (bf16 A-fragments)
  {
    const int nWhh = G3H_ * H_;
    pack_b_w<<<(nWhh + 255) / 256, 256, 0, stream>>>(w_hh0, whh0p, G3H_, H_);
    pack_b_w<<<(nWhh + 255) / 256, 256, 0, stream>>>(w_hh1, whh1p, G3H_, H_);
    pack_b_w<<<(nWhh + 255) / 256, 256, 0, stream>>>(w_ih1, wih1p, G3H_, H_);
    const int nWih0 = G3H_ * D_;
    pack_b_w<<<(nWih0 + 255) / 256, 256, 0, stream>>>(w_ih0, wih0p, G3H_, D_);
    const int nX = BT_ * D_;
    pack_a_f32<<<(nX + 255) / 256, 256, 0, stream>>>(x, xpk, BT_, D_);
  }

  const dim3 gemmGrid(G3H_ / 16, BT_ / 64);  // 192 x 256, 4 waves/block
  // 2) layer-0 input gates: xg = x @ w_ih0^T
  wmma_gemm<<<gemmGrid, 128, 0, stream>>>(xpk, wih0p, xg, G3H_, D_ / 32);

  // 3) layer-0 recurrence (single persistent launch)
  zero_u32<<<(B_ * H_ + 255) / 256, 256, 0, stream>>>((unsigned int*)hf32, B_ * H_);
  zero_u32<<<(B_ * H_ / 2 + 255) / 256, 256, 0, stream>>>((unsigned int*)hp0, B_ * H_ / 2);
  zero_u32<<<1, 32, 0, stream>>>((unsigned int*)bar, 2);
  gru_layer<<<NBLK, 384, 0, stream>>>(xg, hp0, hp1, hf32, whh0p, b_ih0, b_hh0, h1pk, bar);

  // 4) layer-1 input gates: xg = h1 @ w_ih1^T (reuse xg buffer)
  wmma_gemm<<<gemmGrid, 128, 0, stream>>>(h1pk, wih1p, xg, G3H_, H_ / 32);

  // 5) layer-1 recurrence
  zero_u32<<<(B_ * H_ + 255) / 256, 256, 0, stream>>>((unsigned int*)hf32, B_ * H_);
  zero_u32<<<(B_ * H_ / 2 + 255) / 256, 256, 0, stream>>>((unsigned int*)hp0, B_ * H_ / 2);
  zero_u32<<<1, 32, 0, stream>>>((unsigned int*)bar, 2);
  gru_layer<<<NBLK, 384, 0, stream>>>(xg, hp0, hp1, hf32, whh1p, b_ih1, b_hh1,
                                      (unsigned short*)nullptr, bar);

  // 6) FC head on final hidden state
  fc_kernel<<<B_, 128, 0, stream>>>(hf32, fc_w, fc_b, out);
}
